// VectorQuantizer_50079318671612
// MI455X (gfx1250) — matrix operand order repaired and checked by reading, not compiled, not run on hardware
//
#include <hip/hip_runtime.h>

// ---------------------------------------------------------------------------
// VectorQuantizer for MI455X (gfx1250): fused WMMA GEMM + argmin + gather.
//   hidden: (32768, 64) f32,  codebook: (8192, 64) f32
//   out = [ z_q (2097152 f32) | indices (32768, as f32) | loss (1 f32) ]
// ws layout: [0,1MB) packed f16 B tiles | [1MB,+32KB) e_sq f32 | then 256 partials
//
// Inner loop is software-pipelined: WMMAs for tile t are issued before the
// argmin epilogue of tile t-1, hiding the WMMA->VALU RAW hazard (4 co-exec
// slots) and letting the XDL matrix op co-execute with the select VALU.
// ---------------------------------------------------------------------------

typedef __attribute__((ext_vector_type(16))) _Float16 v16h;
typedef __attribute__((ext_vector_type(8)))  float    v8f;

#define NUM_E   8192
#define DIM     64
#define NTOK    32768
#define NTILE   512              // 8192 codes / 16 per tile
#define SCALE_F 4096.0f          // power-of-2 prescale so f16 codebook stays normal
#define INV_2S  (2.0f / 4096.0f)

// --- Kernel 1: pack codebook into WMMA B layout (f16, prescaled) + e_sq -----
// B (32x16, KxN) per-lane layout used here (contiguous K, same order as A):
//   lanes 0-15  hold N=lane,    K = 0..15  (2 halves per VGPR, K ascending)
//   lanes 16-31 hold N=lane-16, K = 16..31
// Packed element gid = (t*2 + c)*32 + lane holds 16 f16 for dim chunk c.
__global__ __launch_bounds__(256) void vq_pack(const float* __restrict__ emb,
                                               v16h* __restrict__ bpack,
                                               float* __restrict__ esq) {
  int gid  = blockIdx.x * 256 + threadIdx.x;   // 0..32767
  int lane = gid & 31;
  int c    = (gid >> 5) & 1;                   // dim chunk (0..31 / 32..63)
  int t    = gid >> 6;                         // code tile
  int col  = lane & 15;                        // N within tile
  int hi   = lane >> 4;                        // K half
  const float* row = emb + (size_t)(t * 16 + col) * DIM + c * 32 + hi * 16;
  v16h v;
#pragma unroll
  for (int j = 0; j < 16; ++j) v[j] = (_Float16)(row[j] * SCALE_F);
  bpack[gid] = v;

  if (gid < NUM_E) {                           // exact f32 ||e_k||^2
    const float* er = emb + (size_t)gid * DIM;
    float s = 0.f;
#pragma unroll 8
    for (int d = 0; d < DIM; ++d) s = fmaf(er[d], er[d], s);
    esq[gid] = s;
  }
}

// --- Kernel 2: fused GEMM(argmin) + gather + partial loss -------------------
__global__ __launch_bounds__(256) void vq_main(const float* __restrict__ h,
                                               const float* __restrict__ emb,
                                               const v16h* __restrict__ bpack,
                                               const float* __restrict__ esq,
                                               float* __restrict__ zq_out,
                                               float* __restrict__ idx_out,
                                               float* __restrict__ bsum) {
  const int lane = threadIdx.x & 31;
  const int wave = threadIdx.x >> 5;
  const int tokenBase = blockIdx.x * 128 + wave * 16;  // 16 tokens per wave
  const int hi  = lane >> 4;      // lane half (K split for A, M split for C)
  const int col = lane & 15;

  // Build A (16x32 f16, contiguous-K layout, same as B) for both dim chunks, once.
  // lanes 0-15: M=lane, v[0..15] <- K {0..15}
  // lanes 16-31: same M, K {16..31}
  const float* hrow = h + (size_t)(tokenBase + col) * DIM;
  v16h a0, a1;
#pragma unroll
  for (int j = 0; j < 8; ++j) {
    a0[j]     = (_Float16)hrow[hi * 16 + j];
    a0[j + 8] = (_Float16)hrow[hi * 16 + 8 + j];
    a1[j]     = (_Float16)hrow[32 + hi * 16 + j];
    a1[j + 8] = (_Float16)hrow[32 + hi * 16 + 8 + j];
  }

  float best[8];
  int   bidx[8];
#pragma unroll
  for (int r = 0; r < 8; ++r) { best[r] = 3.4e38f; bidx[r] = 0x7fffffff; }

  // ---- software-pipelined GEMM + argmin over 512 code tiles ----
  // Prologue: tile 0 accumulators.
  v8f accP = {};
  {
    v16h b0 = bpack[lane];
    v16h b1 = bpack[32 + lane];
    accP = __builtin_amdgcn_wmma_f32_16x16x32_f16(false, a0, false, b0,
                                                  (short)0, accP, false, false);
    accP = __builtin_amdgcn_wmma_f32_16x16x32_f16(false, a1, false, b1,
                                                  (short)0, accP, false, false);
  }

#pragma unroll 2
  for (int t = 1; t < NTILE; ++t) {
    // prefetch 4 tiles (8 KB) ahead, keep in near caches (global_prefetch_b8)
    __builtin_prefetch(
        (const void*)&bpack[(((t + 4) & (NTILE - 1)) * 2) * 32 + lane], 0, 3);

    // Issue tile t loads + WMMAs first (no dependency on accP epilogue).
    v16h b0 = bpack[(t * 2 + 0) * 32 + lane];
    v16h b1 = bpack[(t * 2 + 1) * 32 + lane];
    v8f acc = {};
    acc = __builtin_amdgcn_wmma_f32_16x16x32_f16(false, a0, false, b0,
                                                 (short)0, acc, false, false);
    acc = __builtin_amdgcn_wmma_f32_16x16x32_f16(false, a1, false, b1,
                                                 (short)0, acc, false, false);

    // Epilogue for tile t-1 co-executes with the matrix op above.
    // accP[r] = 4096 * (z_m . e_n), m = r + hi*8, this lane's n = (t-1)*16+col
    const float eq  = esq[(t - 1) * 16 + col];
    const int   idx = (t - 1) * 16 + col;
#pragma unroll
    for (int r = 0; r < 8; ++r) {
      float s = fmaf(-INV_2S, accP[r], eq);   // ||e||^2 - 2 z.e
      if (s < best[r]) { best[r] = s; bidx[r] = idx; }
    }
    accP = acc;
  }
  { // drain: epilogue for the last tile
    const float eq  = esq[(NTILE - 1) * 16 + col];
    const int   idx = (NTILE - 1) * 16 + col;
#pragma unroll
    for (int r = 0; r < 8; ++r) {
      float s = fmaf(-INV_2S, accP[r], eq);
      if (s < best[r]) { best[r] = s; bidx[r] = idx; }
    }
  }

  // Reduce (score, idx) across the 16 lanes sharing each token row.
  // Tie-break: smaller index wins (matches jnp.argmin first-occurrence).
#pragma unroll
  for (int off = 8; off >= 1; off >>= 1) {
#pragma unroll
    for (int r = 0; r < 8; ++r) {
      float os = __shfl_xor(best[r], off, 32);
      int   oi = __shfl_xor(bidx[r], off, 32);
      if (os < best[r] || (os == best[r] && oi < bidx[r])) {
        best[r] = os; bidx[r] = oi;
      }
    }
  }

  if (col == 0) {                      // lane 0 -> tokens 0..7, lane 16 -> 8..15
#pragma unroll
    for (int r = 0; r < 8; ++r)
      idx_out[tokenBase + hi * 8 + r] = (float)bidx[r];   // exact (< 2^24)
  }

  // Gather winning codebook rows (z_q == quantized) + accumulate (q-h)^2.
  float lsum = 0.f;
#pragma unroll 4
  for (int m = 0; m < 16; ++m) {
    int gi = __shfl(bidx[m & 7], (m < 8) ? 0 : 16, 32);
    int R  = tokenBase + m;
    float2 ev = *(const float2*)(emb + (size_t)gi * DIM + lane * 2);
    float2 hv = *(const float2*)(h + (size_t)R * DIM + lane * 2);
    *(float2*)(zq_out + (size_t)R * DIM + lane * 2) = ev;
    float d0 = ev.x - hv.x, d1 = ev.y - hv.y;
    lsum = fmaf(d0, d0, lsum);
    lsum = fmaf(d1, d1, lsum);
  }

  // Deterministic block reduction of the loss partials.
  __shared__ float red[256];
  red[threadIdx.x] = lsum;
  __syncthreads();
#pragma unroll
  for (int s = 128; s > 0; s >>= 1) {
    if (threadIdx.x < s) red[threadIdx.x] += red[threadIdx.x + s];
    __syncthreads();
  }
  if (threadIdx.x == 0) bsum[blockIdx.x] = red[0];
}

// --- Kernel 3: final deterministic loss reduction ---------------------------
__global__ __launch_bounds__(256) void vq_finalize(const float* __restrict__ bsum,
                                                   float* __restrict__ loss) {
  __shared__ float red[256];
  red[threadIdx.x] = bsum[threadIdx.x];
  __syncthreads();
#pragma unroll
  for (int s = 128; s > 0; s >>= 1) {
    if (threadIdx.x < s) red[threadIdx.x] += red[threadIdx.x + s];
    __syncthreads();
  }
  // loss = (1 + 0.25) * mean((q - h)^2) over 2097152 elements
  if (threadIdx.x == 0) loss[0] = red[0] * (1.25f / 2097152.0f);
}

extern "C" void kernel_launch(void* const* d_in, const int* in_sizes, int n_in,
                              void* d_out, int out_size, void* d_ws, size_t ws_size,
                              hipStream_t stream) {
  const float* h   = (const float*)d_in[0];   // (32,32,32,64) f32
  const float* emb = (const float*)d_in[1];   // (8192,64) f32
  float* out  = (float*)d_out;
  float* zq   = out;                          // 2097152
  float* idx  = out + 2097152;                // 32768
  float* loss = out + 2097152 + 32768;        // 1

  char*  ws    = (char*)d_ws;
  v16h*  bpack = (v16h*)ws;                         // 1 MB packed f16 B tiles
  float* esq   = (float*)(ws + (1 << 20));          // 32 KB ||e||^2
  float* bsum  = (float*)(ws + (1 << 20) + (32 << 10)); // 1 KB partials

  vq_pack    <<<NUM_E * 4 / 256, 256, 0, stream>>>(emb, bpack, esq);
  vq_main    <<<NTOK / 128,      256, 0, stream>>>(h, emb, bpack, esq, zq, idx, bsum);
  vq_finalize<<<1,               256, 0, stream>>>(bsum, loss);
}
